// CommonModule_15805479649350
// MI455X (gfx1250) — matrix operand order repaired and checked by reading, not compiled, not run on hardware
//
#include <hip/hip_runtime.h>
#include <hip/hip_bf16.h>

typedef __attribute__((ext_vector_type(16))) _Float16 v16h;
typedef __attribute__((ext_vector_type(8)))  float    v8f;

#define NBATCH 8
#define SEQ    1024
#define UNIF   128
#define NH     8
#define HD     16

// softmax scale 1/sqrt(16)=0.25 folded with log2(e) so we can use exp2
#define QSCALE 0.36067376022224085f

__device__ __forceinline__ float fast_exp2(float x) {
#if __has_builtin(__builtin_amdgcn_exp2f)
  return __builtin_amdgcn_exp2f(x);       // v_exp_f32 (TRANS, co-executes w/ WMMA)
#else
  return exp2f(x);
#endif
}

// DPP16 butterfly max over the 16 lanes of each half-row (pure VALU, no DS).
// Float-typed update_dpp keeps value provenance -> DPP folds into v_max_num_f32.
template <int CTRL>
__device__ __forceinline__ float dpp_maxf(float x) {
  float y = __builtin_amdgcn_update_dpp(0.0f, x, CTRL, 0xF, 0xF, true);
  return fmaxf(x, y);
}
__device__ __forceinline__ float rowmax16(float x) {
  x = dpp_maxf<0xB1>(x);    // quad_perm [1,0,3,2]  (xor 1)
  x = dpp_maxf<0x4E>(x);    // quad_perm [2,3,0,1]  (xor 2)
  x = dpp_maxf<0x141>(x);   // row_half_mirror      (covers the 8-group)
  x = dpp_maxf<0x140>(x);   // row_mirror           (covers the 16-group)
  return x;
}

// ---------------------------------------------------------------------------
// Kernel 1 (templated on Z): proj = vec[8192,128] @ W[128,128]  (f32 in, f16 out)
// Z==0 -> q stored [b,n,h,d]; Z==1 -> k stored [b,h,n,d]; Z==2 -> v stored [b,h,d,n]
// ---------------------------------------------------------------------------
template <int Z>
__global__ __launch_bounds__(256)
void proj_kernel(const float* __restrict__ vec,
                 const float* __restrict__ W,
                 _Float16* __restrict__ outp) {
  __shared__ _Float16 wT[128][136];              // transposed f16 weight, padded

  const int wave = threadIdx.x >> 5;
  const int lane = threadIdx.x & 31;
  const int m    = lane & 15;
  const int kb   = (lane >> 4) * 8;              // unused; A frag also uses kb2
  const int kb2  = (lane >> 4) * 16;             // B-frag K base for this half
  const int row0 = blockIdx.x * 128 + wave * 16;

  // A fragments for K = 0..127 (f32 -> f16); issued before the barrier so the
  // global loads overlap the LDS weight staging.
  v16h a[4];
  #pragma unroll
  for (int kk = 0; kk < 4; ++kk) {
    const float* src = vec + (size_t)(row0 + m) * 128 + kk * 32;
    #pragma unroll
    for (int e = 0; e < 8; ++e)  a[kk][e] = (_Float16)src[kb2 + e];
    #pragma unroll
    for (int e = 8; e < 16; ++e) a[kk][e] = (_Float16)src[kb2 + e];
  }

  for (int idx = threadIdx.x; idx < 128 * 128; idx += 256) {
    int x = idx >> 7, c = idx & 127;
    wT[c][x] = (_Float16)W[idx];
  }
  __syncthreads();

  #pragma unroll
  for (int nt = 0; nt < 8; ++nt) {
    v8f acc = {};
    #pragma unroll
    for (int kk = 0; kk < 4; ++kk) {
      v16h b;
      const _Float16* wp = &wT[nt * 16 + m][kk * 32 + kb2];
      #pragma unroll
      for (int e = 0; e < 16; ++e) b[e] = wp[e];
      acc = __builtin_amdgcn_wmma_f32_16x16x32_f16(false, a[kk], false, b,
                                                   (short)0, acc, false, false);
    }
    #pragma unroll
    for (int r = 0; r < 8; ++r) {
      int row = row0 + r + ((lane >> 4) << 3);
      int col = nt * 16 + m;
      _Float16 hv = (_Float16)acc[r];
      if (Z == 0) {
        outp[(size_t)row * 128 + col] = hv;                         // [b,n,h,d]
      } else {
        int b_ = row >> 10, n_ = row & 1023;
        int h_ = col >> 4,  d_ = col & 15;
        if (Z == 1) {
          outp[(((size_t)(b_ * NH + h_) * SEQ) + n_) * HD + d_] = hv;   // [b,h,n,d]
        } else {
          outp[(((size_t)(b_ * NH + h_) * HD) + d_) * SEQ + n_] = hv;   // [b,h,d,n]
        }
      }
    }
  }
}

// ---------------------------------------------------------------------------
// Kernel 2: flash attention, one wave per (b, h, 16-query tile).
// Row max:  DPP butterfly (VALU).    Row sum:  P @ ones via one extra WMMA.
// ---------------------------------------------------------------------------
__global__ __launch_bounds__(128)
void attention_kernel(const _Float16* __restrict__ q,    // [b,n,h,d]
                      const _Float16* __restrict__ k,    // [b,h,n,d]
                      const _Float16* __restrict__ v,    // [b,h,d,n]
                      _Float16* __restrict__ recv) {     // [b,n,h,d]
  __shared__ _Float16 plds[4][16][40];                   // 80B row stride (16B aligned)
  const int wave = threadIdx.x >> 5;
  const int lane = threadIdx.x & 31;
  const int m    = lane & 15;
  const int half = lane >> 4;
  const int kb   = half * 8;
  const int kb2  = half * 16;

  const int b_ = blockIdx.z;
  const int h_ = blockIdx.y;
  const int i0 = (blockIdx.x * 4 + wave) * 16;

  // Q A-fragment: K-dim padded 16->32 with zeros; scale folded in.
  const _Float16* qp = q + (((size_t)(b_ * SEQ + i0 + m) * NH) + h_) * HD;
  v16h qa = {};
  if (half == 0) {
  #pragma unroll
  for (int e = 0; e < 16; ++e)                           // K = e in [0,16); half 1 -> K>=16 -> 0
    qa[e] = (_Float16)((float)qp[e] * QSCALE);
  }

  v16h ones;
  #pragma unroll
  for (int e = 0; e < 16; ++e) ones[e] = (_Float16)1.0f;

  const _Float16* kp0 = k + (size_t)(b_ * NH + h_) * SEQ * HD;   // [n][d]
  const _Float16* vp0 = v + (size_t)(b_ * NH + h_) * HD * SEQ    // [d][n]
                          + (size_t)m * SEQ + kb2;

  v8f o = {};          // output accumulator  (rows x d)
  v8f lacc = {};       // row-sum accumulator (every column = row sum)
  float mst[8];
  #pragma unroll
  for (int r = 0; r < 8; ++r) mst[r] = -1e30f;

  for (int j0 = 0; j0 < SEQ; j0 += 32) {
    // ---- S = Q K^T over 32 keys (two wmmas, K-dim = d padded) ----
    v16h kf0 = {}, kf1 = {};
    if (half == 0) {                                     // upper half = K>=16 pad
      const _Float16* kp = kp0 + (size_t)(j0 + m) * HD;  // 32B contiguous
      #pragma unroll
      for (int e = 0; e < 16; ++e) kf0[e] = kp[e];
      const _Float16* kp2 = kp + 16 * HD;
      #pragma unroll
      for (int e = 0; e < 16; ++e) kf1[e] = kp2[e];
    }
    v8f zc = {};
    v8f s0 = __builtin_amdgcn_wmma_f32_16x16x32_f16(false, qa, false, kf0, (short)0, zc, false, false);
    v8f s1 = __builtin_amdgcn_wmma_f32_16x16x32_f16(false, qa, false, kf1, (short)0, zc, false, false);

    // ---- V B-fragment: 16 contiguous f16 per lane -> two b128 loads ----
    v16h vf;
    const _Float16* vp = vp0 + j0;
    #pragma unroll
    for (int e = 0; e < 16; ++e) vf[e] = vp[e];

    // ---- online softmax: DPP row max, exp2, rescale accumulators ----
    v8f p0, p1;
    #pragma unroll
    for (int r = 0; r < 8; ++r) {
      float x0 = s0[r], x1 = s1[r];
      float mx = rowmax16(fmaxf(x0, x1));
      float mnew  = fmaxf(mst[r], mx);
      float alpha = fast_exp2(mst[r] - mnew);
      p0[r] = fast_exp2(x0 - mnew);
      p1[r] = fast_exp2(x1 - mnew);
      o[r]    *= alpha;
      lacc[r] *= alpha;
      mst[r] = mnew;
    }

    // ---- C-layout -> A-layout for P via per-wave LDS tile ----
    __builtin_amdgcn_wave_barrier();
    #pragma unroll
    for (int r = 0; r < 8; ++r) {
      int row = r + (half << 3);
      plds[wave][row][m]      = (_Float16)p0[r];
      plds[wave][row][m + 16] = (_Float16)p1[r];
    }
    __builtin_amdgcn_wave_barrier();
    asm volatile("s_wait_dscnt 0" ::: "memory");

    v16h pa;
    #pragma unroll
    for (int e = 0; e < 8; ++e)  pa[e] = plds[wave][m][kb2 + e];
    #pragma unroll
    for (int e = 8; e < 16; ++e) pa[e] = plds[wave][m][kb2 + e];
    __builtin_amdgcn_wave_barrier();

    // ---- O += P V ; L += P @ ones (row sums land in every lane) ----
    o    = __builtin_amdgcn_wmma_f32_16x16x32_f16(false, pa, false, vf,   (short)0, o,    false, false);
    lacc = __builtin_amdgcn_wmma_f32_16x16x32_f16(false, pa, false, ones, (short)0, lacc, false, false);
  }

  // ---- normalize, store recv [b,n,h,d] as f16 ----
  #pragma unroll
  for (int r = 0; r < 8; ++r) {
    float rinv = 1.0f / lacc[r];
    int row = i0 + r + (half << 3);
    size_t idx = (((size_t)(b_ * SEQ + row) * NH) + h_) * HD + m;
    recv[idx] = (_Float16)(o[r] * rinv);
  }
}

// ---------------------------------------------------------------------------
// Kernel 3: out[8192,128] = recv[8192,128](f16) @ Wo[128,128]  (f32 out)
// ---------------------------------------------------------------------------
__global__ __launch_bounds__(256)
void out_proj_kernel(const _Float16* __restrict__ recv,
                     const float* __restrict__ Wo,
                     float* __restrict__ out) {
  __shared__ _Float16 wT[128][136];

  const int wave = threadIdx.x >> 5;
  const int lane = threadIdx.x & 31;
  const int m    = lane & 15;
  const int kb   = (lane >> 4) * 8;
  const int kb2  = (lane >> 4) * 16;
  const int row0 = blockIdx.x * 128 + wave * 16;

  v16h a[4];
  #pragma unroll
  for (int kk = 0; kk < 4; ++kk) {
    const _Float16* src = recv + (size_t)(row0 + m) * 128 + kk * 32;
    #pragma unroll
    for (int e = 0; e < 8; ++e)  a[kk][e] = src[kb2 + e];
    #pragma unroll
    for (int e = 8; e < 16; ++e) a[kk][e] = src[kb2 + e];
  }

  for (int idx = threadIdx.x; idx < 128 * 128; idx += 256) {
    int x = idx >> 7, c = idx & 127;
    wT[c][x] = (_Float16)Wo[idx];
  }
  __syncthreads();

  #pragma unroll
  for (int nt = 0; nt < 8; ++nt) {
    v8f acc = {};
    #pragma unroll
    for (int kk = 0; kk < 4; ++kk) {
      v16h bfr;
      const _Float16* wp = &wT[nt * 16 + m][kk * 32 + kb2];
      #pragma unroll
      for (int e = 0; e < 16; ++e) bfr[e] = wp[e];
      acc = __builtin_amdgcn_wmma_f32_16x16x32_f16(false, a[kk], false, bfr,
                                                   (short)0, acc, false, false);
    }
    #pragma unroll
    for (int r = 0; r < 8; ++r) {
      int row = row0 + r + ((lane >> 4) << 3);
      int col = nt * 16 + m;
      out[(size_t)row * 128 + col] = acc[r];
    }
  }
}

// ---------------------------------------------------------------------------
extern "C" void kernel_launch(void* const* d_in, const int* in_sizes, int n_in,
                              void* d_out, int out_size, void* d_ws, size_t ws_size,
                              hipStream_t stream) {
  const float* Wq  = (const float*)d_in[0];
  const float* Wk  = (const float*)d_in[1];
  const float* Wv  = (const float*)d_in[2];
  const float* Wo  = (const float*)d_in[3];
  const float* vec = (const float*)d_in[4];

  const size_t M = (size_t)NBATCH * SEQ;       // 8192 rows
  _Float16* q    = (_Float16*)d_ws;            // [b,n,h,d]   2 MB
  _Float16* kbuf = q    + M * UNIF;            // [b,h,n,d]   2 MB
  _Float16* vbuf = kbuf + M * UNIF;            // [b,h,d,n]   2 MB
  _Float16* rcv  = vbuf + M * UNIF;            // [b,n,h,d]   2 MB

  proj_kernel<0><<<dim3(M / 128), 256, 0, stream>>>(vec, Wq, q);
  proj_kernel<1><<<dim3(M / 128), 256, 0, stream>>>(vec, Wk, kbuf);
  proj_kernel<2><<<dim3(M / 128), 256, 0, stream>>>(vec, Wv, vbuf);
  attention_kernel<<<dim3(SEQ / 64, NH, NBATCH), 128, 0, stream>>>(q, kbuf, vbuf, rcv);
  out_proj_kernel<<<dim3(M / 128), 256, 0, stream>>>(rcv, Wo, (float*)d_out);
}
